// EncoderLayer_59339268161894
// MI455X (gfx1250) — matrix operand order repaired and checked by reading, not compiled, not run on hardware
//
#include <hip/hip_runtime.h>
#include <math.h>

// Flip to 0 if the assembler rejects the async-LDS mnemonic.
#define USE_ASYNC_LDS 1

// ---------- constants ----------
static constexpr int Bb  = 4;
static constexpr int Ss  = 2048;
static constexpr int Dd  = 1024;
static constexpr int Hh  = 16;
static constexpr int DKk = 64;
static constexpr int DFF = 4096;
static constexpr int MROWS = Bb * Ss;  // 8192

// ---------- vector types ----------
typedef __attribute__((ext_vector_type(16))) __bf16 v16bf;  // 32B, WMMA operand
typedef __attribute__((ext_vector_type(8)))  __bf16 v8bf;   // 16B chunk
typedef __attribute__((ext_vector_type(8)))  float  v8f;

// float -> bf16 (round-to-nearest-even)
static __device__ __forceinline__ __bf16 f2bf(float f) {
  union { float f; unsigned u; } c; c.f = f;
  unsigned r = c.u + 0x7FFFu + ((c.u >> 16) & 1u);
  unsigned short h = (unsigned short)(r >> 16);
  return __builtin_bit_cast(__bf16, h);
}

static __device__ __forceinline__ v16bf pack2(v8bf lo, v8bf hi) {
  v16bf r;
#pragma unroll
  for (int i = 0; i < 8; ++i) { r[i] = lo[i]; r[i + 8] = hi[i]; }
  return r;
}

static __device__ __forceinline__ v8f wmma_bf16(v16bf a, v16bf b, v8f c) {
  return __builtin_amdgcn_wmma_f32_16x16x32_bf16(false, a, false, b, (short)0, c,
                                                 false, false);
}

#if USE_ASYNC_LDS
// GLOBAL_LOAD_ASYNC_TO_LDS_B128: per-lane 16B global -> LDS, tracked by ASYNCcnt.
static __device__ __forceinline__ void async_g2l_b128(unsigned lds_byte_addr,
                                                      const void* gaddr) {
  asm volatile("global_load_async_to_lds_b128 %0, %1, off"
               :
               : "v"(lds_byte_addr), "v"(gaddr)
               : "memory");
}
static __device__ __forceinline__ void wait_asynccnt0() {
  asm volatile("s_wait_asynccnt 0" ::: "memory");
}
#endif

// ---------- weight transpose + f32->bf16 convert:  Wt[n][k] = W[k][n] ----------
__global__ __launch_bounds__(256) void convert_transpose_k(
    const float* __restrict__ W, __bf16* __restrict__ Wt, int K, int N) {
  __shared__ float tile[32][33];
  int n0 = blockIdx.x * 32, k0 = blockIdx.y * 32;
  int tx = threadIdx.x & 31, ty = threadIdx.x >> 5;  // 32 x 8
#pragma unroll
  for (int i = 0; i < 4; ++i)
    tile[ty + i * 8][tx] = W[(size_t)(k0 + ty + i * 8) * N + n0 + tx];
  __syncthreads();
#pragma unroll
  for (int i = 0; i < 4; ++i)
    Wt[(size_t)(n0 + ty + i * 8) * K + k0 + tx] = f2bf(tile[tx][ty + i * 8]);
}

// ---------- LayerNorm (f32 in, bf16 out), one row (D=1024) per 256-thread block ----------
__global__ __launch_bounds__(256) void layernorm_k(
    const float* __restrict__ x, const float* __restrict__ w,
    const float* __restrict__ b, __bf16* __restrict__ out) {
  __shared__ float red[2][8];
  int row = blockIdx.x;
  int t = threadIdx.x;
  const float* xr = x + (size_t)row * Dd;
  float4 xv = *(const float4*)&xr[t * 4];
  float s  = xv.x + xv.y + xv.z + xv.w;
  float ss = xv.x * xv.x + xv.y * xv.y + xv.z * xv.z + xv.w * xv.w;
#pragma unroll
  for (int off = 16; off; off >>= 1) {
    s  += __shfl_xor(s, off, 32);
    ss += __shfl_xor(ss, off, 32);
  }
  int lane = t & 31, wid = t >> 5;
  if (lane == 0) { red[0][wid] = s; red[1][wid] = ss; }
  __syncthreads();
  float S1 = 0.f, S2 = 0.f;
#pragma unroll
  for (int i = 0; i < 8; ++i) { S1 += red[0][i]; S2 += red[1][i]; }
  float mu  = S1 * (1.0f / Dd);
  float var = S2 * (1.0f / Dd) - mu * mu;
  float inv = rsqrtf(var + 1e-5f);
  float4 wv = *(const float4*)&w[t * 4];
  float4 bv = *(const float4*)&b[t * 4];
  __bf16* o = out + (size_t)row * Dd + t * 4;
  o[0] = f2bf((xv.x - mu) * inv * wv.x + bv.x);
  o[1] = f2bf((xv.y - mu) * inv * wv.y + bv.y);
  o[2] = f2bf((xv.z - mu) * inv * wv.z + bv.z);
  o[3] = f2bf((xv.w - mu) * inv * wv.w + bv.w);
}

// ---------- Tiled WMMA GEMM: C = A(MxK) @ Bt^T,  Bt is N-major [N][K] bf16 ----------
// Block tile 128x128, 8 waves (wave = 32x64 = 2x4 WMMA tiles), k-step 32.
// Double-buffered LDS filled by async global->LDS; one barrier per k-step.
// MODE 0: fused QKV epilogue; MODE 1: bias+ReLU->bf16; MODE 2: bias+residual->f32.
template <int MODE>
__global__ __launch_bounds__(256) void gemm_bf16_k(
    const __bf16* __restrict__ A, const __bf16* __restrict__ Bt,
    int M, int N, int K,
    const float* __restrict__ bias0, const float* __restrict__ bias1,
    const float* __restrict__ bias2, const float* __restrict__ resid,
    float* __restrict__ outf,
    __bf16* __restrict__ ob0, __bf16* __restrict__ ob1, __bf16* __restrict__ ob2) {
  __shared__ __bf16 As[2][128 * 32];
  __shared__ __bf16 Bs[2][128 * 32];

  int m0 = blockIdx.y * 128;
  int n0 = blockIdx.x * 128;
  int lane = threadIdx.x & 31, wid = threadIdx.x >> 5;
  int wm = (wid & 3) * 32;   // wave M offset in tile
  int wn = (wid >> 2) * 64;  // wave N offset in tile
  int col = lane & 15, half = lane >> 4;

  // this thread's two 16B chunks of the 128x32 tile
  int c0 = threadIdx.x * 2;
  int row_a = c0 >> 2, cc_a = (c0 & 3) * 8;        // chunk c0
  int row_b = (c0 + 1) >> 2, cc_b = ((c0 + 1) & 3) * 8;  // chunk c0+1

  auto stage = [&](int buf, int k0) {
#if USE_ASYNC_LDS
    async_g2l_b128((unsigned)(uintptr_t)&As[buf][row_a * 32 + cc_a],
                   &A[(size_t)(m0 + row_a) * K + k0 + cc_a]);
    async_g2l_b128((unsigned)(uintptr_t)&As[buf][row_b * 32 + cc_b],
                   &A[(size_t)(m0 + row_b) * K + k0 + cc_b]);
    async_g2l_b128((unsigned)(uintptr_t)&Bs[buf][row_a * 32 + cc_a],
                   &Bt[(size_t)(n0 + row_a) * K + k0 + cc_a]);
    async_g2l_b128((unsigned)(uintptr_t)&Bs[buf][row_b * 32 + cc_b],
                   &Bt[(size_t)(n0 + row_b) * K + k0 + cc_b]);
#else
    *(v8bf*)&As[buf][row_a * 32 + cc_a] =
        *(const v8bf*)&A[(size_t)(m0 + row_a) * K + k0 + cc_a];
    *(v8bf*)&As[buf][row_b * 32 + cc_b] =
        *(const v8bf*)&A[(size_t)(m0 + row_b) * K + k0 + cc_b];
    *(v8bf*)&Bs[buf][row_a * 32 + cc_a] =
        *(const v8bf*)&Bt[(size_t)(n0 + row_a) * K + k0 + cc_a];
    *(v8bf*)&Bs[buf][row_b * 32 + cc_b] =
        *(const v8bf*)&Bt[(size_t)(n0 + row_b) * K + k0 + cc_b];
#endif
  };

  v8f acc[2][4];
#pragma unroll
  for (int mt = 0; mt < 2; ++mt)
#pragma unroll
    for (int nt = 0; nt < 4; ++nt) acc[mt][nt] = {};

  int nk = K / 32;
  stage(0, 0);
  for (int it = 0; it < nk; ++it) {
    int cur = it & 1;
#if USE_ASYNC_LDS
    wait_asynccnt0();   // this wave's fills of buf `cur` are done
#endif
    __syncthreads();    // everyone's fills done; prior reads of buf cur^1 done
    if (it + 1 < nk) stage(cur ^ 1, (it + 1) * 32);  // overlap with compute

    v16bf af[2], bfr[4];
#pragma unroll
    for (int mt = 0; mt < 2; ++mt) {
      int row = wm + mt * 16 + col, base = half * 8;
      af[mt] = pack2(*(const v8bf*)&As[cur][row * 32 + base],
                     *(const v8bf*)&As[cur][row * 32 + base + 16]);
    }
#pragma unroll
    for (int nt = 0; nt < 4; ++nt) {
      int nrow = wn + nt * 16 + col, kb = half * 8;
      bfr[nt] = pack2(*(const v8bf*)&Bs[cur][nrow * 32 + kb],
                      *(const v8bf*)&Bs[cur][nrow * 32 + kb + 16]);  // two 16B chunks
    }
#pragma unroll
    for (int mt = 0; mt < 2; ++mt)
#pragma unroll
      for (int nt = 0; nt < 4; ++nt)
        acc[mt][nt] = wmma_bf16(af[mt], bfr[nt], acc[mt][nt]);
  }

  // epilogue: C-frag element (r): row = r + 8*half, col = lane&15
#pragma unroll
  for (int mt = 0; mt < 2; ++mt) {
#pragma unroll
    for (int nt = 0; nt < 4; ++nt) {
#pragma unroll
      for (int r = 0; r < 8; ++r) {
        int m = m0 + wm + mt * 16 + r + 8 * half;
        int n = n0 + wn + nt * 16 + col;
        float v = acc[mt][nt][r];
        if (MODE == 0) {
          int which = n >> 10, d = n & 1023;
          int h = d >> 6, dk = d & 63, bb = m >> 11, sidx = m & 2047;
          const float* bp = (which == 0) ? bias0 : ((which == 1) ? bias1 : bias2);
          v += bp[d];
          if (which == 2) {
            ob2[(size_t)(((bb * Hh + h) * DKk + dk)) * Ss + sidx] = f2bf(v);
          } else {
            __bf16* o = (which == 0) ? ob0 : ob1;
            o[(size_t)(((bb * Hh + h) * Ss + sidx)) * DKk + dk] = f2bf(v);
          }
        } else if (MODE == 1) {
          v += bias0[n];
          v = v > 0.f ? v : 0.f;
          ob0[(size_t)m * N + n] = f2bf(v);
        } else {
          v += bias0[n] + resid[(size_t)m * N + n];
          outf[(size_t)m * N + n] = v;
        }
      }
    }
  }
}

// ---------- Flash attention ----------
// grid (S/128, H, B), 256 threads = 8 waves; each wave owns 16 query rows.
__global__ __launch_bounds__(256) void flash_attn_k(
    const __bf16* __restrict__ q, const __bf16* __restrict__ kk,
    const __bf16* __restrict__ vt, const int* __restrict__ mask,
    __bf16* __restrict__ attn) {
  __shared__ __bf16 P[8 * 16 * 64];  // per-wave 16x64 bf16 P tile

  int lane = threadIdx.x & 31, wid = threadIdx.x >> 5;
  int col = lane & 15, half = lane >> 4;
  int b = blockIdx.z, h = blockIdx.y;
  int q0 = blockIdx.x * 128 + wid * 16;

  const __bf16* qb = q  + (size_t)(b * Hh + h) * Ss * DKk;
  const __bf16* kb = kk + (size_t)(b * Hh + h) * Ss * DKk;
  const __bf16* vb = vt + (size_t)(b * Hh + h) * DKk * Ss;
  __bf16* Pw = &P[wid * 16 * 64];

  // Q A-fragments for k-steps 0,1 (dk 0..31, 32..63)
  v16bf qf[2];
  {
    int qrow = q0 + col;
    int abase = half * 8;
#pragma unroll
    for (int ks = 0; ks < 2; ++ks) {
      const __bf16* p0 = &qb[(size_t)qrow * DKk + ks * 32 + abase];
      qf[ks] = pack2(*(const v8bf*)p0, *(const v8bf*)(p0 + 16));
    }
  }

  v8f O[4];
#pragma unroll
  for (int nt = 0; nt < 4; ++nt) O[nt] = {};
  float mrow[8], lrow[8];
#pragma unroll
  for (int r = 0; r < 8; ++r) { mrow[r] = -1e30f; lrow[r] = 0.f; }
  const float scale = 0.125f;  // 1/sqrt(64)

  for (int kb0 = 0; kb0 < Ss; kb0 += 64) {
    // prefetch next key block's K/V streams into GL2
    if (kb0 + 64 < Ss) {
      __builtin_prefetch(&kb[(size_t)(kb0 + 64 + col * 4) * DKk], 0, 1);
      __builtin_prefetch(&vb[(size_t)(col * 4) * Ss + kb0 + 64], 0, 1);
    }
    float sc[4][8];
#pragma unroll
    for (int nt = 0; nt < 4; ++nt) {
      v8f s = {};
      int key = kb0 + nt * 16 + col;
      int dkb = half * 8;
#pragma unroll
      for (int ks = 0; ks < 2; ++ks) {
        // two 16B chunks (k = 8h+s, 16+8h+s) matching the A fragment
        const __bf16* kp = &kb[(size_t)key * DKk + ks * 32 + dkb];
        v16bf bf = pack2(*(const v8bf*)kp, *(const v8bf*)(kp + 16));
        s = wmma_bf16(qf[ks], bf, s);
      }
      int mv = mask[b * Ss + key];
#pragma unroll
      for (int r = 0; r < 8; ++r) sc[nt][r] = mv ? s[r] * scale : -1e9f;
    }
    // row max across the 4 n-tiles then across the 16 lanes of each half
    float mnew[8];
#pragma unroll
    for (int r = 0; r < 8; ++r) {
      float mx = mrow[r];
#pragma unroll
      for (int nt = 0; nt < 4; ++nt) mx = fmaxf(mx, sc[nt][r]);
#pragma unroll
      for (int off = 8; off; off >>= 1) mx = fmaxf(mx, __shfl_xor(mx, off, 32));
      mnew[r] = mx;
    }
#pragma unroll
    for (int r = 0; r < 8; ++r) {
      float corr = __expf(mrow[r] - mnew[r]);
      lrow[r] *= corr;
      mrow[r] = mnew[r];
#pragma unroll
      for (int nt = 0; nt < 4; ++nt) O[nt][r] *= corr;
    }
    // P = exp(score - m), accumulate partial row sums, stage to LDS (C layout)
#pragma unroll
    for (int nt = 0; nt < 4; ++nt)
#pragma unroll
      for (int r = 0; r < 8; ++r) {
        float p = __expf(sc[nt][r] - mnew[r]);
        lrow[r] += p;
        Pw[(r + 8 * half) * 64 + nt * 16 + col] = f2bf(p);
      }
    // re-load P as A-fragments (same-wave LDS ops are in-order) and do P @ V^T
#pragma unroll
    for (int ks = 0; ks < 2; ++ks) {
      int pb = ks * 32 + half * 8;
      const __bf16* pp = &Pw[col * 64 + pb];
      v16bf pf = pack2(*(const v8bf*)pp, *(const v8bf*)(pp + 16));
#pragma unroll
      for (int nt = 0; nt < 4; ++nt) {
        int dk = nt * 16 + col;
        int keyb = kb0 + ks * 32 + half * 8;
        const __bf16* vp = &vb[(size_t)dk * Ss + keyb];
        v16bf vf = pack2(*(const v8bf*)vp, *(const v8bf*)(vp + 16));  // two 16B chunks
        O[nt] = wmma_bf16(pf, vf, O[nt]);
      }
    }
  }

  // finalize: reduce partial row sums across the half, divide, store bshd bf16
#pragma unroll
  for (int r = 0; r < 8; ++r) {
    float t = lrow[r];
#pragma unroll
    for (int off = 8; off; off >>= 1) t += __shfl_xor(t, off, 32);
    lrow[r] = (t > 0.f) ? (1.0f / t) : 0.f;
  }
#pragma unroll
  for (int nt = 0; nt < 4; ++nt)
#pragma unroll
    for (int r = 0; r < 8; ++r) {
      int row = q0 + r + 8 * half;
      int c = h * DKk + nt * 16 + col;
      attn[(size_t)(b * Ss + row) * Dd + c] = f2bf(O[nt][r] * lrow[r]);
    }
}

// ---------- launch ----------
extern "C" void kernel_launch(void* const* d_in, const int* in_sizes, int n_in,
                              void* d_out, int out_size, void* d_ws, size_t ws_size,
                              hipStream_t stream) {
  const float* x    = (const float*)d_in[0];
  const int*   mask = (const int*)d_in[1];
  const float* wq = (const float*)d_in[2];  const float* bq = (const float*)d_in[3];
  const float* wk = (const float*)d_in[4];  const float* bk = (const float*)d_in[5];
  const float* wv = (const float*)d_in[6];  const float* bv = (const float*)d_in[7];
  const float* wo = (const float*)d_in[8];  const float* bo = (const float*)d_in[9];
  const float* ln1w = (const float*)d_in[10]; const float* ln1b = (const float*)d_in[11];
  const float* ln2w = (const float*)d_in[12]; const float* ln2b = (const float*)d_in[13];
  const float* w1 = (const float*)d_in[14]; const float* b1 = (const float*)d_in[15];
  const float* w2 = (const float*)d_in[16]; const float* b2 = (const float*)d_in[17];

  char* ws = (char*)d_ws;
  size_t off = 0;
  auto alloc = [&](size_t bytes) -> void* {
    void* p = ws + off;
    off += (bytes + 255) & ~(size_t)255;
    return p;
  };
  __bf16* wqkv_t = (__bf16*)alloc((size_t)3 * Dd * Dd * 2);   // [3D][D]
  __bf16* wo_t   = (__bf16*)alloc((size_t)Dd * Dd * 2);       // [D][D]
  __bf16* w1_t   = (__bf16*)alloc((size_t)DFF * Dd * 2);      // [DFF][D]
  __bf16* w2_t   = (__bf16*)alloc((size_t)Dd * DFF * 2);      // [D][DFF]
  __bf16* nx     = (__bf16*)alloc((size_t)MROWS * Dd * 2);
  __bf16* qbuf   = (__bf16*)alloc((size_t)MROWS * Dd * 2);    // [b,h,s,dk]
  __bf16* kbuf   = (__bf16*)alloc((size_t)MROWS * Dd * 2);    // [b,h,s,dk]
  __bf16* vtbuf  = (__bf16*)alloc((size_t)MROWS * Dd * 2);    // [b,h,dk,s]
  __bf16* attnb  = (__bf16*)alloc((size_t)MROWS * Dd * 2);    // [b,s,d]
  float*  x1     = (float*) alloc((size_t)MROWS * Dd * 4);
  __bf16* nx2    = (__bf16*)alloc((size_t)MROWS * Dd * 2);
  __bf16* ffh    = (__bf16*)alloc((size_t)MROWS * DFF * 2);
  (void)in_sizes; (void)n_in; (void)out_size; (void)ws_size;

  // 1) weight prep: transpose + convert to bf16, N-major
  convert_transpose_k<<<dim3(Dd / 32, Dd / 32), 256, 0, stream>>>(wq, wqkv_t, Dd, Dd);
  convert_transpose_k<<<dim3(Dd / 32, Dd / 32), 256, 0, stream>>>(wk, wqkv_t + (size_t)Dd * Dd, Dd, Dd);
  convert_transpose_k<<<dim3(Dd / 32, Dd / 32), 256, 0, stream>>>(wv, wqkv_t + (size_t)2 * Dd * Dd, Dd, Dd);
  convert_transpose_k<<<dim3(Dd / 32, Dd / 32), 256, 0, stream>>>(wo, wo_t, Dd, Dd);
  convert_transpose_k<<<dim3(DFF / 32, Dd / 32), 256, 0, stream>>>(w1, w1_t, Dd, DFF);
  convert_transpose_k<<<dim3(Dd / 32, DFF / 32), 256, 0, stream>>>(w2, w2_t, DFF, Dd);

  // 2) LN1 -> bf16
  layernorm_k<<<MROWS, 256, 0, stream>>>(x, ln1w, ln1b, nx);

  // 3) fused QKV projection (N = 3D), scatter epilogue
  gemm_bf16_k<0><<<dim3(3 * Dd / 128, MROWS / 128), 256, 0, stream>>>(
      nx, wqkv_t, MROWS, 3 * Dd, Dd, bq, bk, bv, nullptr, nullptr,
      qbuf, kbuf, vtbuf);

  // 4) flash attention -> attn (b,s,d) bf16
  flash_attn_k<<<dim3(Ss / 128, Hh, Bb), 256, 0, stream>>>(qbuf, kbuf, vtbuf, mask, attnb);

  // 5) out projection + residual -> x1 (f32)
  gemm_bf16_k<2><<<dim3(Dd / 128, MROWS / 128), 256, 0, stream>>>(
      attnb, wo_t, MROWS, Dd, Dd, bo, nullptr, nullptr, x, x1,
      nullptr, nullptr, nullptr);

  // 6) LN2 -> bf16
  layernorm_k<<<MROWS, 256, 0, stream>>>(x1, ln2w, ln2b, nx2);

  // 7) FFN1 + ReLU -> bf16
  gemm_bf16_k<1><<<dim3(DFF / 128, MROWS / 128), 256, 0, stream>>>(
      nx2, w1_t, MROWS, DFF, Dd, b1, nullptr, nullptr, nullptr, nullptr,
      ffh, nullptr, nullptr);

  // 8) FFN2 + residual -> d_out (f32)
  gemm_bf16_k<2><<<dim3(Dd / 128, MROWS / 128), 256, 0, stream>>>(
      ffh, w2_t, MROWS, Dd, DFF, b2, nullptr, nullptr, x1, (float*)d_out,
      nullptr, nullptr, nullptr);
}
